// LazyAttentionEmbedding_18219251270187
// MI455X (gfx1250) — matrix operand order repaired and checked by reading, not compiled, not run on hardware
//
#include <hip/hip_runtime.h>

// ---------------------------------------------------------------------------
// LazyAttentionEmbedding for MI455X (gfx1250, wave32, WMMA f16->f32)
// B=2 H=16 L=2048 D=64, fp32 in/out, f16 WMMA with f32 accumulate.
// ---------------------------------------------------------------------------

typedef _Float16 v16h __attribute__((ext_vector_type(16)));
typedef _Float16 v8h  __attribute__((ext_vector_type(8)));
typedef _Float16 v4h  __attribute__((ext_vector_type(4)));
typedef float    v8f  __attribute__((ext_vector_type(8)));

#define BATCH   2
#define NHEAD   16
#define SEQ     2048
#define DHEAD   64
#define NQT     (SEQ / 64)      // 32 query tiles of 64 rows
#define LK      72              // LDS row stride in halfs (144B, conflict-free)
#define SCALE   0.125f          // D^-0.5
#define NEG_BIG (-1e30f)

__device__ __forceinline__ v16h ld16c(const _Float16* p) {
  // 16 contiguous halfs -> two b128 reads
  v8h lo = *(const v8h*)(p);
  v8h hi = *(const v8h*)(p + 8);
  v16h r;
#pragma unroll
  for (int i = 0; i < 8; ++i) { r[i] = lo[i]; r[i + 8] = hi[i]; }
  return r;
}

__device__ __forceinline__ v16h ld8p8(const _Float16* p) {
  // A-matrix per-lane pattern: 8 halfs at p, 8 halfs at p+16
  v8h lo = *(const v8h*)(p);
  v8h hi = *(const v8h*)(p + 16);
  v16h r;
#pragma unroll
  for (int i = 0; i < 8; ++i) { r[i] = lo[i]; r[i + 8] = hi[i]; }
  return r;
}

__device__ __forceinline__ float redmax16(float x) {
#pragma unroll
  for (int m = 1; m < 16; m <<= 1) x = fmaxf(x, __shfl_xor(x, m, 32));
  return x;
}
__device__ __forceinline__ float redsum16(float x) {
#pragma unroll
  for (int m = 1; m < 16; m <<= 1) x += __shfl_xor(x, m, 32);
  return x;
}

// Suffix sums of V rows per (b,h): sufv[bh][t][d] = sum_{k >= 64*t} V[k][d]
__global__ void lazyattn_suffix(const float* __restrict__ Vg,
                                float* __restrict__ sufv) {
  const int bh = blockIdx.x;       // 32 blocks
  const int d  = threadIdx.x;      // 64 threads
  const float* Vb = Vg + (size_t)bh * SEQ * DHEAD;
  float* S = sufv + (size_t)bh * (NQT + 1) * DHEAD;
  float acc = 0.f;
  S[NQT * DHEAD + d] = 0.f;
  for (int t = NQT - 1; t >= 0; --t) {
    const float* base = Vb + (size_t)t * 64 * DHEAD;
#pragma unroll 4
    for (int k = 0; k < 64; ++k) acc += base[k * DHEAD + d];
    S[t * DHEAD + d] = acc;
  }
}

__global__ __launch_bounds__(128)
void lazyattn_main(const float* __restrict__ Qg, const float* __restrict__ Kg,
                   const float* __restrict__ Vg, const float* __restrict__ taug,
                   const float* __restrict__ biasg,
                   const float* __restrict__ sufv, float* __restrict__ Og) {
  __shared__ __align__(16) _Float16 sK [64 * LK];      // K tile  [key][d]   f16
  __shared__ __align__(16) _Float16 sVt[64 * LK];      // V tile  [d][key]   f16
  __shared__ __align__(16) _Float16 sP [4][16 * LK];   // per-wave P scratch
  __shared__ float sB[128];                            // bias diagonal band
  __shared__ float sVS[DHEAD];                         // tail V colsum (fallback)

  const int tid  = threadIdx.x;
  const int lane = tid & 31;
  const int wv   = tid >> 5;       // wave 0..3
  const int n    = lane & 15;      // column index inside C/B layouts
  const int hi   = lane >> 4;      // half-wave selector

  const int bid = blockIdx.x;      // 1024 = 2*16*32
  const int qt  = bid & (NQT - 1);
  const int h   = (bid >> 5) & (NHEAD - 1);
  const int b   = bid >> 9;
  const int q0  = qt * 64;

  const float tau = taug[h];
  const size_t bh = (size_t)(b * NHEAD + h);
  const float* Qb = Qg + bh * (size_t)(SEQ * DHEAD);
  const float* Kb = Kg + bh * (size_t)(SEQ * DHEAD);
  const float* Vb = Vg + bh * (size_t)(SEQ * DHEAD);
  float*       Ob = Og + bh * (size_t)(SEQ * DHEAD);

  // ---- Q tile rows for this wave, directly in WMMA A layout (f32 -> f16)
  v16h aQ[2];
  {
    const float* qrow = Qb + (size_t)(q0 + wv * 16 + n) * DHEAD;
#pragma unroll
    for (int c = 0; c < 2; ++c) {
      const int d0 = 32 * c + 8 * hi;       // elems 0..7
      const int d1 = 32 * c + 16 + 8 * hi;  // elems 8..15
      float4 f0 = *(const float4*)(qrow + d0);
      float4 f1 = *(const float4*)(qrow + d0 + 4);
      float4 f2 = *(const float4*)(qrow + d1);
      float4 f3 = *(const float4*)(qrow + d1 + 4);
      v16h a;
      a[0]  = (_Float16)f0.x; a[1]  = (_Float16)f0.y;
      a[2]  = (_Float16)f0.z; a[3]  = (_Float16)f0.w;
      a[4]  = (_Float16)f1.x; a[5]  = (_Float16)f1.y;
      a[6]  = (_Float16)f1.z; a[7]  = (_Float16)f1.w;
      a[8]  = (_Float16)f2.x; a[9]  = (_Float16)f2.y;
      a[10] = (_Float16)f2.z; a[11] = (_Float16)f2.w;
      a[12] = (_Float16)f3.x; a[13] = (_Float16)f3.y;
      a[14] = (_Float16)f3.z; a[15] = (_Float16)f3.w;
      aQ[c] = a;
    }
  }

  const int qlocw = 16 * wv + 8 * hi;   // local q row = qlocw + j
  const int nkt   = qt + 1;             // causal key tiles: k0 = 0..q0

  // ---- cooperative staging of one 64-key tile (K always; V on demand)
  auto stageTile = [&](int k0, bool withV) {
    const int kr = tid >> 1, ch = tid & 1;
    {
      const float* krow = Kb + (size_t)(k0 + kr) * DHEAD + 32 * ch;
      _Float16* dst = sK + kr * LK + 32 * ch;
#pragma unroll
      for (int i = 0; i < 8; ++i) {
        float4 f = *(const float4*)(krow + 4 * i);
        v4h hh;
        hh[0] = (_Float16)f.x; hh[1] = (_Float16)f.y;
        hh[2] = (_Float16)f.z; hh[3] = (_Float16)f.w;
        *(v4h*)(dst + 4 * i) = hh;
      }
    }
    if (withV) {
      const float* vrow = Vb + (size_t)(k0 + kr) * DHEAD + 32 * ch;
#pragma unroll
      for (int i = 0; i < 8; ++i) {
        float4 f = *(const float4*)(vrow + 4 * i);
        const int d = 32 * ch + 4 * i;
        sVt[(d + 0) * LK + kr] = (_Float16)f.x;
        sVt[(d + 1) * LK + kr] = (_Float16)f.y;
        sVt[(d + 2) * LK + kr] = (_Float16)f.z;
        sVt[(d + 3) * LK + kr] = (_Float16)f.w;
      }
    }
    // bias band: rel = q - k spans [q0-k0-63, q0-k0+63]
    int rel = q0 - k0 - 63 + tid;
    int idx = rel < 0 ? 0 : (rel > 4095 ? 4095 : rel);
    sB[tid] = biasg[idx * NHEAD + h];
  };

  // ---- S = Q * K^T for one tile (8 WMMAs), result in C layout
  auto computeS = [&](v8f S[4]) {
#pragma unroll
    for (int t = 0; t < 4; ++t) { v8f z = {}; S[t] = z; }
#pragma unroll
    for (int c = 0; c < 2; ++c) {
#pragma unroll
      for (int t = 0; t < 4; ++t) {
        v16h bK = ld8p8(&sK[(16 * t + n) * LK + 32 * c + 8 * hi]);
        S[t] = __builtin_amdgcn_wmma_f32_16x16x32_f16(
            false, aQ[c], false, bK, (short)0, S[t], false, false);
      }
    }
  };

  // =========================== PASS 1: stats ===============================
  float mrow[8], zrow[8];
#pragma unroll
  for (int j = 0; j < 8; ++j) { mrow[j] = NEG_BIG; zrow[j] = 0.f; }

  for (int it = 0; it < nkt; ++it) {
    const int k0 = it * 64;
    __syncthreads();
    stageTile(k0, false);
    __syncthreads();

    v8f S[4];
    computeS(S);
    const bool needMask = (k0 + 63) > (q0 + 16 * wv);
#pragma unroll
    for (int j = 0; j < 8; ++j) {
      const int qloc = qlocw + j;
      float sv[4];
      float mx = NEG_BIG;
#pragma unroll
      for (int t = 0; t < 4; ++t) {
        const int kloc = 16 * t + n;
        float s = S[t][j] * SCALE + sB[qloc - kloc + 63];
        if (needMask && (k0 + kloc) > (q0 + qloc)) s = NEG_BIG;
        sv[t] = s;
        mx = fmaxf(mx, s);
      }
      mx = redmax16(mx);
      const float mnew = fmaxf(mrow[j], mx);
      float ssum = 0.f;
#pragma unroll
      for (int t = 0; t < 4; ++t) ssum += __expf(sv[t] - mnew);
      ssum = redsum16(ssum);
      zrow[j] = zrow[j] * __expf(mrow[j] - mnew) + ssum;
      mrow[j] = mnew;
    }
  }

  float invz[8], cj[8];
#pragma unroll
  for (int j = 0; j < 8; ++j) {
    invz[j] = 1.f / zrow[j];
    cj[j]   = tau / (float)(q0 + qlocw + j + 1);
  }

  // =========================== PASS 2: P @ V ===============================
  v8f O[4];
#pragma unroll
  for (int t = 0; t < 4; ++t) { v8f z = {}; O[t] = z; }

  for (int it = 0; it < nkt; ++it) {
    const int k0 = it * 64;
    __syncthreads();
    stageTile(k0, true);
    __syncthreads();

    v8f S[4];
    computeS(S);
    const bool needMask = (k0 + 63) > (q0 + 16 * wv);
    // weights: w = relu(exp(s-m)/Z + tau/(q+1)); masked -> relu(c) automatically
#pragma unroll
    for (int j = 0; j < 8; ++j) {
      const int qloc = qlocw + j;
#pragma unroll
      for (int t = 0; t < 4; ++t) {
        const int kloc = 16 * t + n;
        float s = S[t][j] * SCALE + sB[qloc - kloc + 63];
        if (needMask && (k0 + kloc) > (q0 + qloc)) s = NEG_BIG;
        float p = __expf(s - mrow[j]) * invz[j];
        float w = fmaxf(p + cj[j], 0.f);
        sP[wv][(j + 8 * hi) * LK + kloc] = (_Float16)w;
      }
    }
    // P (C layout in LDS) -> A layout, then 8 WMMAs of P @ V
#pragma unroll
    for (int c = 0; c < 2; ++c) {
      v16h aP = ld8p8(&sP[wv][n * LK + 32 * c + 8 * hi]);
#pragma unroll
      for (int t = 0; t < 4; ++t) {
        v16h bV = ld8p8(&sVt[(16 * t + n) * LK + 32 * c + 8 * hi]);
        O[t] = __builtin_amdgcn_wmma_f32_16x16x32_f16(
            false, aP, false, bV, (short)0, O[t], false, false);
      }
    }
  }

  // ================== tau > 0 tail: relu(c) * suffix-sum(V) ================
  const int ktail = q0 + 64;
  const bool dotail = (tau > 0.f) && (ktail < SEQ);
  if (dotail && sufv == nullptr) {          // fallback: in-block column sum
    __syncthreads();
    if (tid < DHEAD) {
      float acc = 0.f;
      for (int k = ktail; k < SEQ; ++k) acc += Vb[(size_t)k * DHEAD + tid];
      sVS[tid] = acc;
    }
    __syncthreads();
  }
  if (dotail) {
    float suf[4];
#pragma unroll
    for (int t = 0; t < 4; ++t) {
      const int d = 16 * t + n;
      suf[t] = sufv ? sufv[(bh * (NQT + 1) + (ktail >> 6)) * DHEAD + d]
                    : sVS[d];
    }
#pragma unroll
    for (int j = 0; j < 8; ++j) {
      const float w = cj[j];                // tau > 0 => relu(c) == c
#pragma unroll
      for (int t = 0; t < 4; ++t) O[t][j] += w * suf[t];
    }
  }

  // ============================== store out ================================
#pragma unroll
  for (int t = 0; t < 4; ++t)
#pragma unroll
    for (int j = 0; j < 8; ++j) {
      const int r = q0 + 16 * wv + j + 8 * hi;
      Ob[(size_t)r * DHEAD + 16 * t + n] = O[t][j];
    }
}

extern "C" void kernel_launch(void* const* d_in, const int* in_sizes, int n_in,
                              void* d_out, int out_size, void* d_ws,
                              size_t ws_size, hipStream_t stream) {
  (void)in_sizes; (void)n_in; (void)out_size;
  const float* q    = (const float*)d_in[0];
  const float* k    = (const float*)d_in[1];
  const float* v    = (const float*)d_in[2];
  const float* tau  = (const float*)d_in[3];
  const float* bias = (const float*)d_in[4];
  // d_in[5] = max_bias_length (4096); L=2048 so clamp to [0,4095] is exact.

  const size_t sufBytes = (size_t)BATCH * NHEAD * (NQT + 1) * DHEAD * sizeof(float);
  const float* sufv = nullptr;
  if (ws_size >= sufBytes) {
    lazyattn_suffix<<<BATCH * NHEAD, DHEAD, 0, stream>>>(v, (float*)d_ws);
    sufv = (const float*)d_ws;
  }
  lazyattn_main<<<BATCH * NHEAD * NQT, 128, 0, stream>>>(
      q, k, v, tau, bias, sufv, (float*)d_out);
}
